// ConditionalAttention_17806934409721
// MI455X (gfx1250) — matrix operand order repaired and checked by reading, not compiled, not run on hardware
//
#include <hip/hip_runtime.h>

// ---------------------------------------------------------------------------
// Types for CDNA5 WMMA (wave32, 16x16x32 bf16 -> f32)
// ---------------------------------------------------------------------------
typedef __bf16 bf16_t;
typedef __attribute__((ext_vector_type(8)))  __bf16 v8bf;
typedef __attribute__((ext_vector_type(16))) __bf16 v16bf;
typedef __attribute__((ext_vector_type(8)))  float  v8f;

// Problem constants
#define BATCH 4
#define NTOK  2048
#define DMODEL 512
#define NHEAD 8
#define DHEAD 64

// ---------------------------------------------------------------------------
// Wave32 lane-shuffle reductions via ds_swizzle (group-of-32: lane ^ mask)
// imm = {xor[14:10], or[9:5], and[4:0]} ; and=0x1f, or=0, xor=mask
// ---------------------------------------------------------------------------
#define SWZ_F(v, m) __int_as_float(__builtin_amdgcn_ds_swizzle(__float_as_int(v), (((m) << 10) | 0x1f)))

__device__ __forceinline__ float red_max16(float v) {
  v = fmaxf(v, SWZ_F(v, 1));
  v = fmaxf(v, SWZ_F(v, 2));
  v = fmaxf(v, SWZ_F(v, 4));
  v = fmaxf(v, SWZ_F(v, 8));
  return v;
}
__device__ __forceinline__ float red_sum16(float v) {
  v += SWZ_F(v, 1);
  v += SWZ_F(v, 2);
  v += SWZ_F(v, 4);
  v += SWZ_F(v, 8);
  return v;
}
__device__ __forceinline__ float red_sum32(float v) {
  v += SWZ_F(v, 1);
  v += SWZ_F(v, 2);
  v += SWZ_F(v, 4);
  v += SWZ_F(v, 8);
  v += SWZ_F(v, 16);
  return v;
}

// ---------------------------------------------------------------------------
// WMMA fragment loaders (wave32; A and B both use contiguous k order k = hi*16+e)
//
// A fragment (16x32 bf16): lane L: m = L&15, hi = L>>4
//   elements 0..7  = A[m][hi*16   .. hi*16+7]
//   elements 8..15 = A[m][hi*16+8 .. hi*16+15]
// B fragment (32x16 bf16): lane L: n = L&15, hi = L>>4
//   elements e     = B[hi*16+e][n]   (contiguous if B is stored K-major per column)
// C/D (16x16 f32): lane L: element r -> row hi*8+r, col = L&15
// ---------------------------------------------------------------------------
__device__ __forceinline__ v16bf load_a_frag(const bf16_t* base, int ld, int ln16, int hi) {
  const bf16_t* p0 = base + (size_t)ln16 * ld + hi * 16;
  v8bf x0 = *(const v8bf*)p0;
  v8bf x1 = *(const v8bf*)(p0 + 8);
  v16bf r;
#pragma unroll
  for (int i = 0; i < 8; ++i) { r[i] = x0[i]; r[i + 8] = x1[i]; }
  return r;
}

// base points at the "column 0 / k 0" corner; per-column rows are contiguous
// (ld = stride between columns, elements)
__device__ __forceinline__ v16bf load_b_frag(const bf16_t* base, int ld, int ln16, int hi) {
  return *(const v16bf*)(base + (size_t)ln16 * ld + hi * 16);
}

__device__ __forceinline__ v8f wmma_bf16(v16bf a, v16bf b, v8f c) {
  return __builtin_amdgcn_wmma_f32_16x16x32_bf16(false, a, false, b, (short)0, c, false, false);
}

// ---------------------------------------------------------------------------
// Kernel 1: weight transpose + f32 -> bf16  (Wt[n][k] = W[k][n], 512x512)
// ---------------------------------------------------------------------------
__global__ void transpose_convert_kernel(const float* __restrict__ w, bf16_t* __restrict__ wt) {
  int idx = blockIdx.x * blockDim.x + threadIdx.x;
  if (idx >= DMODEL * DMODEL) return;
  int n = idx / DMODEL;
  int k = idx - n * DMODEL;
  wt[idx] = (bf16_t)w[(size_t)k * DMODEL + n];
}

// ---------------------------------------------------------------------------
// Kernel 2: LayerNorm, one wave (32 lanes) per 512-wide row, bf16 output
// ---------------------------------------------------------------------------
__global__ void layernorm_kernel(const float* __restrict__ x, const float* __restrict__ gam,
                                 const float* __restrict__ bet, bf16_t* __restrict__ xn) {
  int tid = threadIdx.x;
  int w = tid >> 5;
  int lane = tid & 31;
  size_t row = (size_t)blockIdx.x * 8 + w;
  const float* xr = x + row * DMODEL;

  float vals[16];
#pragma unroll
  for (int i = 0; i < 4; ++i) {
    float4 t = *(const float4*)(xr + lane * 16 + i * 4);
    vals[i * 4 + 0] = t.x; vals[i * 4 + 1] = t.y;
    vals[i * 4 + 2] = t.z; vals[i * 4 + 3] = t.w;
  }
  float s = 0.f;
#pragma unroll
  for (int i = 0; i < 16; ++i) s += vals[i];
  s = red_sum32(s);
  float mu = s * (1.0f / DMODEL);
  float vs = 0.f;
#pragma unroll
  for (int i = 0; i < 16; ++i) { float d = vals[i] - mu; vs += d * d; }
  vs = red_sum32(vs);
  float rstd = rsqrtf(vs * (1.0f / DMODEL) + 1e-5f);

  bf16_t* out = xn + row * DMODEL + lane * 16;
#pragma unroll
  for (int i = 0; i < 16; ++i) {
    int c = lane * 16 + i;
    out[i] = (bf16_t)((vals[i] - mu) * rstd * gam[c] + bet[c]);
  }
}

// ---------------------------------------------------------------------------
// Kernel 3: generic bf16 WMMA GEMM  C[M x 512] = A[M x 512] * Wt^T + bias
//   A   : bf16 row-major [M x 512]
//   Bt  : bf16 [512 x 512], Bt[n][k] = W[k][n]  (column-K-major)
//   MODE 0: out bf16 row-major,   val = (acc + bias)*alpha
//   MODE 1: out f32  row-major
//   MODE 2: out bf16 transposed per head: Vt[b][h][d][n_tok]
// Block: 256 threads = 8 waves; wave -> 16(M) x 64(N) tile; grid (N/64, M/128)
// K-loop is software-pipelined (next A prefetched) and partially unrolled so
// the scheduler can interleave global_load clauses with WMMAs.
// ---------------------------------------------------------------------------
template <int MODE>
__global__ void gemm_bf16_kernel(const bf16_t* __restrict__ A, const bf16_t* __restrict__ Bt,
                                 const float* __restrict__ bias, void* __restrict__ outp,
                                 float alpha) {
  constexpr int K = DMODEL, NC = DMODEL;
  int tid = threadIdx.x;
  int w = tid >> 5;
  int lane = tid & 31;
  int ln16 = lane & 15;
  int hi = lane >> 4;
  int m0 = blockIdx.y * 128 + w * 16;
  int n0 = blockIdx.x * 64;

  v8f acc[4] = {v8f{}, v8f{}, v8f{}, v8f{}};

  const bf16_t* arow = A + (size_t)m0 * K;

  // Prologue: prefetch first A fragment
  v16bf a_cur = load_a_frag(arow, K, ln16, hi);

#pragma unroll 4
  for (int kk = 0; kk < K; kk += 32) {
    // Prefetch next A fragment while current WMMAs run
    v16bf a_next;
    if (kk + 32 < K) a_next = load_a_frag(arow + kk + 32, K, ln16, hi);
#pragma unroll
    for (int nf = 0; nf < 4; ++nf) {
      v16bf b = load_b_frag(Bt + (size_t)(n0 + nf * 16) * K + kk, K, ln16, hi);
      acc[nf] = wmma_bf16(a_cur, b, acc[nf]);
    }
    a_cur = a_next;
  }

#pragma unroll
  for (int nf = 0; nf < 4; ++nf) {
    int col = n0 + nf * 16 + ln16;
    float bv = bias[col];
#pragma unroll
    for (int r = 0; r < 8; ++r) {
      int row = m0 + hi * 8 + r;
      float val = (acc[nf][r] + bv) * alpha;
      if (MODE == 0) {
        ((bf16_t*)outp)[(size_t)row * NC + col] = (bf16_t)val;
      } else if (MODE == 1) {
        ((float*)outp)[(size_t)row * NC + col] = val;
      } else {
        int bb = row >> 11;            // row / NTOK
        int nt = row & (NTOK - 1);
        int hh = col >> 6;
        int d = col & 63;
        ((bf16_t*)outp)[(((size_t)bb * NHEAD + hh) * DHEAD + d) * NTOK + nt] = (bf16_t)val;
      }
    }
  }
}

// ---------------------------------------------------------------------------
// Kernel 4: causal flash attention.
//   q,k : bf16 [B*N][512] (col = h*64+d), q pre-scaled by dk^-0.5
//   vt  : bf16 [B][H][64][N]
//   o   : bf16 [B*N][512]
// Block = 256 thr = 8 waves, each wave owns 16 query rows; grid = B*H*(N/128)
// ---------------------------------------------------------------------------
__global__ void attention_kernel(const bf16_t* __restrict__ q, const bf16_t* __restrict__ k,
                                 const bf16_t* __restrict__ vt, bf16_t* __restrict__ o) {
  __shared__ __align__(16) bf16_t pbuf[8][16][32];

  int tid = threadIdx.x;
  int w = tid >> 5;
  int lane = tid & 31;
  int ln16 = lane & 15;
  int hi = lane >> 4;

  int qblk = blockIdx.x & 15;
  int h = (blockIdx.x >> 4) & 7;
  int b = blockIdx.x >> 7;
  int i0 = qblk * 128 + w * 16;

  const bf16_t* qbase = q + ((size_t)(b * NTOK + i0)) * DMODEL + h * DHEAD;
  const bf16_t* kb = k + (size_t)b * NTOK * DMODEL + h * DHEAD;
  const bf16_t* vtb = vt + ((size_t)(b * NHEAD + h)) * DHEAD * NTOK;

  v16bf aq0 = load_a_frag(qbase, DMODEL, ln16, hi);
  v16bf aq1 = load_a_frag(qbase + 32, DMODEL, ln16, hi);

  v8f accO[4] = {v8f{}, v8f{}, v8f{}, v8f{}};
  float mrow[8], lrow[8];
#pragma unroll
  for (int r = 0; r < 8; ++r) { mrow[r] = -1e30f; lrow[r] = 0.f; }

  int rowbase = i0 + hi * 8;

  for (int j0 = 0; j0 < i0 + 16; j0 += 32) {
    // ---- S = Q * K^T for 32 keys (two 16-col C fragments) ----
    v8f s0 = v8f{}, s1 = v8f{};
    {
      v16bf b00 = load_b_frag(kb + (size_t)j0 * DMODEL, DMODEL, ln16, hi);
      v16bf b01 = load_b_frag(kb + (size_t)j0 * DMODEL + 32, DMODEL, ln16, hi);
      v16bf b10 = load_b_frag(kb + (size_t)(j0 + 16) * DMODEL, DMODEL, ln16, hi);
      v16bf b11 = load_b_frag(kb + (size_t)(j0 + 16) * DMODEL + 32, DMODEL, ln16, hi);
      s0 = wmma_bf16(aq0, b00, s0);
      s0 = wmma_bf16(aq1, b01, s0);
      s1 = wmma_bf16(aq0, b10, s1);
      s1 = wmma_bf16(aq1, b11, s1);
    }

    // ---- causal mask + online softmax (per C-layout row hi*8+r) ----
    float p0[8], p1[8], alf[8];
#pragma unroll
    for (int r = 0; r < 8; ++r) {
      int ig = rowbase + r;
      int jg0 = j0 + ln16;
      int jg1 = j0 + 16 + ln16;
      float a0 = (jg0 > ig) ? -1e30f : s0[r];
      float a1 = (jg1 > ig) ? -1e30f : s1[r];
      float rm = red_max16(fmaxf(a0, a1));
      float mn = fmaxf(mrow[r], rm);
      float sc = __expf(mrow[r] - mn);
      float e0 = __expf(a0 - mn);
      float e1 = __expf(a1 - mn);
      float rs = red_sum16(e0 + e1);
      lrow[r] = lrow[r] * sc + rs;
      mrow[r] = mn;
      alf[r] = sc;
      p0[r] = e0;
      p1[r] = e1;
    }
#pragma unroll
    for (int nf = 0; nf < 4; ++nf)
#pragma unroll
      for (int r = 0; r < 8; ++r) accO[nf][r] *= alf[r];

    // ---- P (C layout) -> LDS -> A fragment (16x32) ----
#pragma unroll
    for (int r = 0; r < 8; ++r) {
      pbuf[w][hi * 8 + r][ln16] = (bf16_t)p0[r];
      pbuf[w][hi * 8 + r][16 + ln16] = (bf16_t)p1[r];
    }
    asm volatile("s_wait_dscnt 0x0" ::: "memory");
    v16bf ap = load_a_frag((const bf16_t*)&pbuf[w][0][0], 32, ln16, hi);

    // ---- O += P * V  (Vt columns are contiguous tokens) ----
#pragma unroll
    for (int nf = 0; nf < 4; ++nf) {
      v16bf bv = load_b_frag(vtb + (size_t)(nf * 16) * NTOK + j0, NTOK, ln16, hi);
      accO[nf] = wmma_bf16(ap, bv, accO[nf]);
    }
  }

  // ---- epilogue: divide by l, store bf16 ----
  bf16_t* obase = o + ((size_t)(b * NTOK + i0)) * DMODEL + h * DHEAD;
#pragma unroll
  for (int nf = 0; nf < 4; ++nf) {
#pragma unroll
    for (int r = 0; r < 8; ++r) {
      float inv = 1.0f / lrow[r];
      obase[(size_t)(hi * 8 + r) * DMODEL + nf * 16 + ln16] = (bf16_t)(accO[nf][r] * inv);
    }
  }
}

// ---------------------------------------------------------------------------
// Launch: weights->bf16^T, LN, QKV GEMMs, attention, out-proj GEMM
// ---------------------------------------------------------------------------
extern "C" void kernel_launch(void* const* d_in, const int* in_sizes, int n_in,
                              void* d_out, int out_size, void* d_ws, size_t ws_size,
                              hipStream_t stream) {
  (void)in_sizes; (void)n_in; (void)out_size; (void)ws_size;
  const float* x = (const float*)d_in[0];
  // d_in[1] = attn_mask (causal, implemented analytically)
  const float* ln_g = (const float*)d_in[2];
  const float* ln_b = (const float*)d_in[3];
  const float* wq = (const float*)d_in[4];
  const float* bq = (const float*)d_in[5];
  const float* wk = (const float*)d_in[6];
  const float* bk = (const float*)d_in[7];
  const float* wv = (const float*)d_in[8];
  const float* bv = (const float*)d_in[9];
  const float* wo = (const float*)d_in[10];
  const float* bo = (const float*)d_in[11];
  float* out = (float*)d_out;

  const size_t rows = (size_t)BATCH * NTOK;          // 8192
  char* p = (char*)d_ws;
  auto take = [&](size_t bytes) -> void* {
    void* r = (void*)p;
    p += ((bytes + 255) / 256) * 256;
    return r;
  };
  bf16_t* xn  = (bf16_t*)take(rows * DMODEL * sizeof(bf16_t));
  bf16_t* wqT = (bf16_t*)take((size_t)DMODEL * DMODEL * sizeof(bf16_t));
  bf16_t* wkT = (bf16_t*)take((size_t)DMODEL * DMODEL * sizeof(bf16_t));
  bf16_t* wvT = (bf16_t*)take((size_t)DMODEL * DMODEL * sizeof(bf16_t));
  bf16_t* woT = (bf16_t*)take((size_t)DMODEL * DMODEL * sizeof(bf16_t));
  bf16_t* qb  = (bf16_t*)take(rows * DMODEL * sizeof(bf16_t));
  bf16_t* kbuf = (bf16_t*)take(rows * DMODEL * sizeof(bf16_t));
  bf16_t* vtb = (bf16_t*)take(rows * DMODEL * sizeof(bf16_t));
  bf16_t* ao  = (bf16_t*)take(rows * DMODEL * sizeof(bf16_t));

  const int tcBlocks = (DMODEL * DMODEL + 255) / 256;   // 1024
  transpose_convert_kernel<<<tcBlocks, 256, 0, stream>>>(wq, wqT);
  transpose_convert_kernel<<<tcBlocks, 256, 0, stream>>>(wk, wkT);
  transpose_convert_kernel<<<tcBlocks, 256, 0, stream>>>(wv, wvT);
  transpose_convert_kernel<<<tcBlocks, 256, 0, stream>>>(wo, woT);

  layernorm_kernel<<<(int)(rows / 8), 256, 0, stream>>>(x, ln_g, ln_b, xn);

  dim3 ggrid(DMODEL / 64, (unsigned)(rows / 128));      // (8, 64)
  const float qscale = 0.125f;                           // 64^-0.5
  gemm_bf16_kernel<0><<<ggrid, 256, 0, stream>>>(xn, wqT, bq, (void*)qb, qscale);
  gemm_bf16_kernel<0><<<ggrid, 256, 0, stream>>>(xn, wkT, bk, (void*)kbuf, 1.0f);
  gemm_bf16_kernel<2><<<ggrid, 256, 0, stream>>>(xn, wvT, bv, (void*)vtb, 1.0f);

  attention_kernel<<<BATCH * NHEAD * (NTOK / 128), 256, 0, stream>>>(qb, kbuf, vtb, ao);

  gemm_bf16_kernel<1><<<ggrid, 256, 0, stream>>>(ao, woT, bo, (void*)out, 1.0f);
}
